// MaskedMHA_22840636080656
// MI455X (gfx1250) — matrix operand order repaired and checked by reading, not compiled, not run on hardware
//
#include <hip/hip_runtime.h>

typedef __bf16 bf16_t;
typedef __attribute__((ext_vector_type(16))) __bf16          v16bf;
typedef __attribute__((ext_vector_type(8)))  float           v8f;
typedef __attribute__((ext_vector_type(8)))  unsigned int    v8u;
typedef __attribute__((ext_vector_type(16))) unsigned short  v16us;

#define B_  4
#define S_  2048
#define D_  768
#define H_  12
#define DK_ 64

__device__ __forceinline__ v8f wmma_bf16(v16bf a, v16bf b, v8f c) {
  return __builtin_amdgcn_wmma_f32_16x16x32_bf16(
      /*neg_a=*/false, a, /*neg_b=*/false, b,
      /*c_mod=*/(short)0, c, /*reuse_a=*/false, /*reuse_b=*/false);
}

__device__ __forceinline__ unsigned short f2bfbits(float x) {
  __bf16 b = (__bf16)x;
  return __builtin_bit_cast(unsigned short, b);
}

// B-matrix tile (32x16, bf16, K=32): lane holds column N = lane&15,
// packed elements i = consecutive K starting at (lane&16). Source must be
// 16 consecutive bf16 (p already offset by (lane&16) by the caller).
__device__ __forceinline__ v16bf load_row32(const bf16_t* p) {
  const unsigned* u = (const unsigned*)p;
  v8u t;
#pragma unroll
  for (int j = 0; j < 8; ++j) t[j] = u[j];
  return __builtin_bit_cast(v16bf, t);
}

// A-matrix tile (16x32, bf16) from row-major bf16: row_ptr points at this
// lane's row (row = base + (lane&15)); K(i) = (lane>=16?16:0) + i.
// Pairs (2j,2j+1) are consecutive K -> dword loads.
__device__ __forceinline__ v16bf load_a_bf(const bf16_t* row_ptr, int lane, int kbase) {
  v8u t;
  int off = (lane & 16);  // 0 or 16
#pragma unroll
  for (int j = 0; j < 8; ++j) {
    int kk = kbase + off + 2 * j;
    t[j] = *(const unsigned*)(row_ptr + kk);
  }
  return __builtin_bit_cast(v16bf, t);
}

// Same A layout but converting from f32 on the fly.
__device__ __forceinline__ v16bf load_a_f32(const float* row_ptr, int lane, int kbase) {
  v16us h;
  int off = (lane & 16);
#pragma unroll
  for (int j = 0; j < 8; ++j) {
    int kk = kbase + off + 2 * j;
    h[2 * j]     = f2bfbits(row_ptr[kk]);
    h[2 * j + 1] = f2bfbits(row_ptr[kk + 1]);
  }
  return __builtin_bit_cast(v16bf, h);
}

// All-ones bf16 B-matrix (for WMMA row-sum reduction).
__device__ __forceinline__ v16bf ones_bf16() {
  v16us h;
#pragma unroll
  for (int i = 0; i < 16; ++i) h[i] = 0x3f80;  // bf16 1.0
  return __builtin_bit_cast(v16bf, h);
}

// ---------------------------------------------------------------------------
// GEMM + bias: C[8192 x 768] = A[8192 x 768] * W[768 x 768] + bias
// MODE 0/1: A=f32, out bf16 in [B,H,S,DK]   (Q / K projections)
// MODE 2  : A=f32, out bf16 in [B,H,DK,S]   (V projection, transposed)
// MODE 3  : A=bf16, out f32 row-major       (output projection)
// Block: 256 threads = 8 waves; each wave owns 32 rows x 64 cols
// (two A tiles share every staged B tile -> 2x WMMA per LDS read).
// ---------------------------------------------------------------------------
template <int MODE>
__global__ __launch_bounds__(256) void gemm_bias_kernel(
    const float* __restrict__ Af, const bf16_t* __restrict__ Ab,
    const float* __restrict__ W, const float* __restrict__ bias,
    bf16_t* __restrict__ outb, float* __restrict__ outf) {
  __shared__ bf16_t wtile[64 * 32];  // [col][k], bf16

  const int wave = threadIdx.x >> 5;
  const int lane = threadIdx.x & 31;
  const int rowBase = blockIdx.x * 256 + wave * 32;
  const int colBase = blockIdx.y * 64;

  v8f acc[2][4];
#pragma unroll
  for (int r = 0; r < 2; ++r)
#pragma unroll
    for (int t = 0; t < 4; ++t) acc[r][t] = (v8f){};

  for (int kk = 0; kk < D_; kk += 32) {
    __syncthreads();
    // Stage W[kk..kk+31][colBase..colBase+63] into LDS as bf16 [col][k]
#pragma unroll
    for (int e = 0; e < 8; ++e) {
      int idx = threadIdx.x + 256 * e;  // 0..2047
      int k = idx >> 6;
      int c = idx & 63;
      wtile[c * 32 + k] = (bf16_t)W[(size_t)(kk + k) * D_ + colBase + c];
    }
    __syncthreads();

    v16bf a0, a1;
    if (MODE == 3) {
      a0 = load_a_bf(Ab + (size_t)(rowBase + (lane & 15)) * D_, lane, kk);
      a1 = load_a_bf(Ab + (size_t)(rowBase + 16 + (lane & 15)) * D_, lane, kk);
    } else {
      a0 = load_a_f32(Af + (size_t)(rowBase + (lane & 15)) * D_, lane, kk);
      a1 = load_a_f32(Af + (size_t)(rowBase + 16 + (lane & 15)) * D_, lane, kk);
    }

#pragma unroll
    for (int t = 0; t < 4; ++t) {
      const bf16_t* bp = wtile + (t * 16 + (lane & 15)) * 32 + (lane & 16);
      v16bf bmat = load_row32(bp);
      acc[0][t] = wmma_bf16(a0, bmat, acc[0][t]);
      acc[1][t] = wmma_bf16(a1, bmat, acc[1][t]);
    }
  }

  // Epilogue. C layout: M = v + (lane>=16?8:0), N = (lane&15) + 16*t
  const int nn = lane & 15;
  const int half8 = (lane >> 4) << 3;
#pragma unroll
  for (int r = 0; r < 2; ++r) {
#pragma unroll
    for (int v = 0; v < 8; ++v) {
      int m = rowBase + 16 * r + v + half8;
      int bb = m >> 11;       // / S_
      int ss = m & (S_ - 1);
#pragma unroll
      for (int t = 0; t < 4; ++t) {
        int n = colBase + t * 16 + nn;
        float c = acc[r][t][v] + bias[n];
        if (MODE <= 1) {
          int h = n >> 6, dk = n & 63;
          outb[(((size_t)bb * H_ + h) * S_ + ss) * DK_ + dk] = (bf16_t)c;
        } else if (MODE == 2) {
          int h = n >> 6, dk = n & 63;
          outb[(((size_t)bb * H_ + h) * DK_ + dk) * S_ + ss] = (bf16_t)c;
        } else {
          outf[(size_t)m * D_ + n] = c;
        }
      }
    }
  }
}

// ---------------------------------------------------------------------------
// Causal flash attention. One wave per 16-query tile; 4 waves per block.
// Qp/Kp: [B,H,S,DK] bf16 ; Vt: [B,H,DK,S] bf16 ; AO: [B,S,H*DK] bf16.
// Row max uses 16-lane xor-shuffles; row SUM is done on the matrix unit
// (P x ones) which lands replicated in exactly the li[] layout.
// ---------------------------------------------------------------------------
__global__ __launch_bounds__(128) void attn_kernel(
    const bf16_t* __restrict__ Qp, const bf16_t* __restrict__ Kp,
    const bf16_t* __restrict__ Vt, bf16_t* __restrict__ AO) {
  __shared__ bf16_t pbuf[4][16 * 32];  // per-wave P staging (C->A transpose)

  const int wave = threadIdx.x >> 5;
  const int lane = threadIdx.x & 31;
  const int h = blockIdx.y;
  const int b = blockIdx.z;
  const int q0 = (blockIdx.x * 4 + wave) * 16;

  const bf16_t* Qh = Qp + ((size_t)b * H_ + h) * S_ * DK_;
  const bf16_t* Kh = Kp + ((size_t)b * H_ + h) * S_ * DK_;
  const bf16_t* Vh = Vt + ((size_t)b * H_ + h) * DK_ * S_;

  // Q tile, A layout, both 32-wide K chunks of DK=64
  const bf16_t* qrow = Qh + (size_t)(q0 + (lane & 15)) * DK_;
  const v16bf qa0 = load_a_bf(qrow, lane, 0);
  const v16bf qa1 = load_a_bf(qrow, lane, 32);
  const v16bf ones = ones_bf16();

  v8f acc[4];
#pragma unroll
  for (int t = 0; t < 4; ++t) acc[t] = (v8f){};
  float mi[8], li[8];
#pragma unroll
  for (int v = 0; v < 8; ++v) { mi[v] = -__builtin_inff(); li[v] = 0.0f; }

  const int nn = lane & 15;
  const int half8 = (lane >> 4) << 3;
  const int kend = q0 + 16;  // causal: keys <= q0+15

  bf16_t* pb = &pbuf[wave][0];

  for (int k0 = 0; k0 < kend; k0 += 32) {
    // scores: two 16x16 tiles, keys [k0..k0+15] and [k0+16..k0+31]
    const bf16_t* krow1 = Kh + (size_t)(k0 + nn) * DK_ + (lane & 16);
    const bf16_t* krow2 = krow1 + 16 * DK_;
    v8f s1 = (v8f){}, s2 = (v8f){};
    s1 = wmma_bf16(qa0, load_row32(krow1), s1);
    s1 = wmma_bf16(qa1, load_row32(krow1 + 32), s1);
    s2 = wmma_bf16(qa0, load_row32(krow2), s2);
    s2 = wmma_bf16(qa1, load_row32(krow2 + 32), s2);

    // scale + causal mask (element (M,N): q=q0+M, key=k0(+16)+N)
#pragma unroll
    for (int v = 0; v < 8; ++v) {
      int qrowi = q0 + v + half8;
      float x1 = s1[v] * 0.125f;               // 1/sqrt(64)
      float x2 = s2[v] * 0.125f;
      if (k0 + nn > qrowi)      x1 = -__builtin_inff();
      if (k0 + 16 + nn > qrowi) x2 = -__builtin_inff();
      s1[v] = x1; s2[v] = x2;
    }

    // row max via 16-lane xor reduction; exp; rescale accumulators
    float corr[8];
#pragma unroll
    for (int v = 0; v < 8; ++v) {
      float rm = fmaxf(s1[v], s2[v]);
#pragma unroll
      for (int m = 8; m >= 1; m >>= 1) rm = fmaxf(rm, __shfl_xor(rm, m, 32));
      float mn = fmaxf(mi[v], rm);
      corr[v] = __expf(mi[v] - mn);
      mi[v] = mn;
      s1[v] = __expf(s1[v] - mn);
      s2[v] = __expf(s2[v] - mn);
#pragma unroll
      for (int t = 0; t < 4; ++t) acc[t][v] *= corr[v];
    }

    // P (C layout) -> LDS -> reload as A-matrix 16x32 (per-wave buffer;
    // same-wave LDS ops are in-order, wait on dscnt before the reads land)
#pragma unroll
    for (int v = 0; v < 8; ++v) {
      int row = v + half8;
      pb[row * 32 + nn]      = (bf16_t)s1[v];
      pb[row * 32 + 16 + nn] = (bf16_t)s2[v];
    }
    asm volatile("s_wait_dscnt 0" ::: "memory");
    v16bf pa;
    {
      const bf16_t* rp = pb + (lane & 15) * 32;
      v8u t8;
      int off = (lane & 16);
#pragma unroll
      for (int j = 0; j < 8; ++j) {
        int kk = off + 2 * j;
        t8[j] = *(const unsigned*)(rp + kk);
      }
      pa = __builtin_bit_cast(v16bf, t8);
    }
    asm volatile("" ::: "memory");

    // row sums on the matrix unit: (P x ones) -> C[M,N] = rowsum(M),
    // replicated across lanes in the same layout as li[].
    v8f rs = wmma_bf16(pa, ones, (v8f){});
#pragma unroll
    for (int v = 0; v < 8; ++v) li[v] = li[v] * corr[v] + rs[v];

    // O += P x V : V chunk is 32(keys) x 64(dv) = 4 B-tiles, contiguous in Vt
#pragma unroll
    for (int t = 0; t < 4; ++t) {
      const bf16_t* vp = Vh + (size_t)(t * 16 + nn) * S_ + k0 + (lane & 16);
      acc[t] = wmma_bf16(pa, load_row32(vp), acc[t]);
    }
  }

  // normalize + store merged-head layout [b][s][h*64+dv] as bf16
#pragma unroll
  for (int v = 0; v < 8; ++v) {
    int row = q0 + v + half8;
    float inv = 1.0f / li[v];
#pragma unroll
    for (int t = 0; t < 4; ++t) {
      int dv = t * 16 + nn;
      AO[((size_t)b * S_ + row) * D_ + h * DK_ + dv] = (bf16_t)(acc[t][v] * inv);
    }
  }
}

extern "C" void kernel_launch(void* const* d_in, const int* in_sizes, int n_in,
                              void* d_out, int out_size, void* d_ws, size_t ws_size,
                              hipStream_t stream) {
  const float* q   = (const float*)d_in[0];
  const float* k   = (const float*)d_in[1];
  const float* v   = (const float*)d_in[2];
  const float* w_q = (const float*)d_in[3];
  const float* b_q = (const float*)d_in[4];
  const float* w_k = (const float*)d_in[5];
  const float* b_k = (const float*)d_in[6];
  const float* w_v = (const float*)d_in[7];
  const float* b_v = (const float*)d_in[8];
  const float* w_o = (const float*)d_in[9];
  const float* b_o = (const float*)d_in[10];
  float* out = (float*)d_out;

  const size_t per = (size_t)B_ * H_ * S_ * DK_;  // elements
  bf16_t* Qp = (bf16_t*)d_ws;
  bf16_t* Kp = Qp + per;
  bf16_t* Vt = Kp + per;
  bf16_t* AO = Vt + per;

  dim3 g(32, 12), blk(256);
  gemm_bias_kernel<0><<<g, blk, 0, stream>>>(q, nullptr, w_q, b_q, Qp, nullptr);
  gemm_bias_kernel<1><<<g, blk, 0, stream>>>(k, nullptr, w_k, b_k, Kp, nullptr);
  gemm_bias_kernel<2><<<g, blk, 0, stream>>>(v, nullptr, w_v, b_v, Vt, nullptr);
  attn_kernel<<<dim3(S_ / 64, H_, B_), 128, 0, stream>>>(Qp, Kp, Vt, AO);
  gemm_bias_kernel<3><<<g, blk, 0, stream>>>(nullptr, AO, w_o, b_o, nullptr, out);
}
